// CausalSelfAttention_78804059947418
// MI455X (gfx1250) — matrix operand order repaired and checked by reading, not compiled, not run on hardware
//
#include <hip/hip_runtime.h>
#include <hip/hip_bf16.h>

typedef __bf16 bf16;
typedef __attribute__((ext_vector_type(16))) __bf16 v16bf;
typedef __attribute__((ext_vector_type(8)))  float  v8f;

union BF16x16 { uint4 q[2]; v16bf v; };
union BF16x4  { ushort4 u; bf16 h[4]; };
union BF16x8  { uint4 u; bf16 h[8]; };

// Problem constants (fixed by the reference)
#define BB 4
#define SS 2048
#define DD 1024
#define HH 16
#define DHD 64
#define MM (BB*SS)   // 8192

// ---------------------------------------------------------------------------
// Kernel 1: fp32 -> bf16 convert (vectorized)
// ---------------------------------------------------------------------------
__global__ __launch_bounds__(256) void convert_bf16_kernel(
    const float* __restrict__ in, bf16* __restrict__ out, long n)
{
    long i = ((long)blockIdx.x * blockDim.x + threadIdx.x) * 4;
    if (i + 3 < n) {
        float4 f = *(const float4*)(in + i);
        BF16x4 p;
        p.h[0] = (bf16)f.x; p.h[1] = (bf16)f.y;
        p.h[2] = (bf16)f.z; p.h[3] = (bf16)f.w;
        *(ushort4*)(out + i) = p.u;
    }
}

// ---------------------------------------------------------------------------
// Kernel 2: W[K][N] fp32 -> Wt[N][K] bf16 (LDS-tiled transpose)
// ---------------------------------------------------------------------------
__global__ __launch_bounds__(256) void transpose_convert_kernel(
    const float* __restrict__ W, bf16* __restrict__ Wt)
{
    __shared__ float tile[32][33];
    const int bx = blockIdx.x * 32;   // N base
    const int by = blockIdx.y * 32;   // K base
    const int tx = threadIdx.x;       // 0..31
    const int ty = threadIdx.y;       // 0..7
    for (int i = 0; i < 32; i += 8)
        tile[ty + i][tx] = W[(long)(by + ty + i) * DD + bx + tx];
    __syncthreads();
    for (int i = 0; i < 32; i += 8)
        Wt[(long)(bx + ty + i) * DD + by + tx] = (bf16)tile[tx][ty + i];
}

// ---------------------------------------------------------------------------
// Kernel 3/5: bf16 GEMM  Out[M][N] = A[M][K] * Wt[N][K]^T, +bias, *scale
// MODE 0: store bf16 into [B,H,S,Dh] head layout (QKV)
// MODE 1: store fp32 into flat [M][N] (output projection)
// BM=128 BN=128 BK=32, 256 threads = 8 waves (2x4), wave tile 64x32.
// ---------------------------------------------------------------------------
template <int MODE>
__global__ __launch_bounds__(256) void gemm_bf16_kernel(
    const bf16* __restrict__ A, const bf16* __restrict__ Bt,
    const float* __restrict__ bias, void* __restrict__ Out, float scale)
{
    constexpr int BM = 128, BN = 128, BK = 32, PAD = 8;
    __shared__ bf16 As[BM][BK + PAD];   // row stride 80B (16B aligned, bank-spread)
    __shared__ bf16 Bs[BN][BK + PAD];

    const int tid    = threadIdx.x;
    const int wave   = tid >> 5;
    const int lane   = tid & 31;
    const int laneLo = lane & 15;
    const int laneHi = lane >> 4;
    const int waveM  = wave >> 2;       // 0..1 : 64 rows
    const int waveN  = wave & 3;        // 0..3 : 32 cols
    const long m0 = (long)blockIdx.x * BM;
    const int  n0 = blockIdx.y * BN;

    const int ldRow = tid >> 1;
    const int ldCol = (tid & 1) * 16;

    v8f acc[4][2];
    {
        v8f z = {};
        for (int i = 0; i < 4; i++) for (int j = 0; j < 2; j++) acc[i][j] = z;
    }

    for (int kb = 0; kb < DD; kb += BK) {
        __syncthreads();
        {   // cooperative tile loads: one 32B segment per thread per matrix
            const uint4* ga = (const uint4*)(A  + (m0 + ldRow) * DD + kb + ldCol);
            uint4 a0 = ga[0], a1 = ga[1];
            *(uint4*)&As[ldRow][ldCol]     = a0;
            *(uint4*)&As[ldRow][ldCol + 8] = a1;
            const uint4* gb = (const uint4*)(Bt + (long)(n0 + ldRow) * DD + kb + ldCol);
            uint4 b0 = gb[0], b1 = gb[1];
            *(uint4*)&Bs[ldRow][ldCol]     = b0;
            *(uint4*)&Bs[ldRow][ldCol + 8] = b1;
        }
        // prefetch next tiles while this one is consumed
        if (kb + BK < DD) {
            __builtin_prefetch(A  + (m0 + ldRow) * DD + kb + BK + ldCol, 0, 0);
            __builtin_prefetch(Bt + (long)(n0 + ldRow) * DD + kb + BK + ldCol, 0, 0);
        }
        __syncthreads();

        BF16x16 af[4], bfr[2];
        for (int tm = 0; tm < 4; tm++) {            // A frag: lane=row, K halves
            int r = waveM * 64 + tm * 16 + laneLo;
            af[tm].q[0] = *(const uint4*)&As[r][laneHi * 8];
            af[tm].q[1] = *(const uint4*)&As[r][16 + laneHi * 8];
        }
        for (int tn = 0; tn < 2; tn++) {            // B frag: lane=col, K halves
            int c = waveN * 32 + tn * 16 + laneLo;
            bfr[tn].q[0] = *(const uint4*)&Bs[c][laneHi * 8];
            bfr[tn].q[1] = *(const uint4*)&Bs[c][16 + laneHi * 8];
        }
        for (int tm = 0; tm < 4; tm++)
            for (int tn = 0; tn < 2; tn++)
                acc[tm][tn] = __builtin_amdgcn_wmma_f32_16x16x32_bf16(
                    false, af[tm].v, false, bfr[tn].v, (short)0, acc[tm][tn],
                    false, false);
    }

    // Epilogue: C layout -> row = vgpr + laneHi*8, col = laneLo (per 16x16 tile)
    for (int tm = 0; tm < 4; tm++) {
        for (int tn = 0; tn < 2; tn++) {
            int col = n0 + waveN * 32 + tn * 16 + laneLo;
            float bv = bias[col];
            for (int e = 0; e < 8; e++) {
                long row = m0 + waveM * 64 + tm * 16 + laneHi * 8 + e;
                float v = (acc[tm][tn][e] + bv) * scale;
                if (MODE == 0) {
                    int h = col >> 6, dh = col & 63;
                    long b = row >> 11, s = row & (SS - 1);
                    ((bf16*)Out)[(((b * HH + h) * SS) + s) * DHD + dh] = (bf16)v;
                } else {
                    ((float*)Out)[row * DD + col] = v;
                }
            }
        }
    }
}

// ---------------------------------------------------------------------------
// Kernel 4: causal flash attention, bf16 WMMA, online softmax (base-2 domain;
// 1/sqrt(Dh)*log2(e) is pre-folded into Q by the GEMM epilogue scale).
// Grid: (S/128, B*H). Block: 256 threads = 8 waves; wave owns 16 query rows.
// Causal skip uses a readfirstlane'd (SGPR) condition so it lowers to a
// scalar branch: EXEC stays all-1s around every WMMA, as the ISA requires.
// ---------------------------------------------------------------------------
__global__ __launch_bounds__(256) void attn_kernel(
    const bf16* __restrict__ Q, const bf16* __restrict__ K,
    const bf16* __restrict__ V, bf16* __restrict__ Outp)
{
    constexpr int PAD = 8;
    __shared__ bf16 Ks[64][DHD + PAD];       // K tile, row-major [key][dh]
    __shared__ bf16 Vt[DHD][64 + PAD];       // V tile transposed [dh][key]
    __shared__ bf16 Ps[8][16][64 + PAD];     // per-wave P staging [qrow][key]

    const int tid    = threadIdx.x;
    const int wave   = tid >> 5;
    const int lane   = tid & 31;
    const int laneLo = lane & 15;
    const int laneHi = lane >> 4;
    const int qb     = blockIdx.x;           // 0..15
    const int bh     = blockIdx.y;           // 0..63
    const long base  = (long)bh * SS * DHD;
    const int qtile  = qb * 128 + wave * 16;
    const int qtileU = __builtin_amdgcn_readfirstlane(qtile);  // wave-uniform SGPR

    // Q fragments (dh 0..31 and 32..63), A layout straight from global
    BF16x16 qf[2];
    {
        const bf16* qrow = Q + base + (long)(qtile + laneLo) * DHD;
        qf[0].q[0] = *(const uint4*)(qrow + laneHi * 8);
        qf[0].q[1] = *(const uint4*)(qrow + 16 + laneHi * 8);
        qf[1].q[0] = *(const uint4*)(qrow + 32 + laneHi * 8);
        qf[1].q[1] = *(const uint4*)(qrow + 48 + laneHi * 8);
    }

    float m[8], l[8];
    v8f o[4];
    {
        v8f z = {};
        for (int e = 0; e < 8; e++) { m[e] = -3.0e38f; l[e] = 0.f; }
        for (int n = 0; n < 4; n++) o[n] = z;
    }

    const int nkb = qb * 2 + 2;              // kv blocks of 64 needed by this block
    for (int kb = 0; kb < nkb; kb++) {
        const int kv0 = kb * 64;
        __syncthreads();
        // Cooperative loads: K straight, V transposed into LDS
        for (int i = 0; i < 2; i++) {
            int idx = tid + i * 256;         // 512 chunks of 8 bf16
            int row = idx >> 3;
            int col = (idx & 7) * 8;
            uint4 kq = *(const uint4*)(K + base + (long)(kv0 + row) * DHD + col);
            *(uint4*)&Ks[row][col] = kq;
            BF16x8 cv;
            cv.u = *(const uint4*)(V + base + (long)(kv0 + row) * DHD + col);
            for (int j = 0; j < 8; j++) Vt[col + j][row] = cv.h[j];
        }
        // prefetch next K/V tiles (8KB each; 256 threads x 32B spread)
        if (kb + 1 < nkb) {
            const long nbase = base + (long)(kv0 + 64) * DHD;
            __builtin_prefetch(K + nbase + tid * 16, 0, 0);
            __builtin_prefetch(V + nbase + tid * 16, 0, 0);
        }
        __syncthreads();

        if (kv0 <= qtileU + 15) {            // scalar (SGPR) causal skip
            // ---- scores S = Q K^T (4 key tiles x 2 dh-chunks) ----
            v8f s[4];
            for (int t = 0; t < 4; t++) {
                int c = t * 16 + laneLo;
                BF16x16 kf0, kf1;            // B frag: lane=key col, elems=dh
                kf0.q[0] = *(const uint4*)&Ks[c][laneHi * 8];
                kf0.q[1] = *(const uint4*)&Ks[c][16 + laneHi * 8];
                kf1.q[0] = *(const uint4*)&Ks[c][32 + laneHi * 8];
                kf1.q[1] = *(const uint4*)&Ks[c][48 + laneHi * 8];
                v8f z = {};
                z = __builtin_amdgcn_wmma_f32_16x16x32_bf16(
                        false, qf[0].v, false, kf0.v, (short)0, z, false, false);
                z = __builtin_amdgcn_wmma_f32_16x16x32_bf16(
                        false, qf[1].v, false, kf1.v, (short)0, z, false, false);
                s[t] = z;
            }
            // ---- causal mask (C layout: row = e+laneHi*8, key = laneLo) ----
            if (kv0 + 63 > qtileU) {         // scalar guard, VALU mask inside
                for (int t = 0; t < 4; t++) {
                    int key = kv0 + t * 16 + laneLo;
                    for (int e = 0; e < 8; e++) {
                        int qr = qtile + laneHi * 8 + e;
                        if (key > qr) s[t][e] = -3.0e38f;
                    }
                }
            }
            // ---- online softmax stats (row reduce across 16-lane halves) ----
            float alpha[8];
            for (int e = 0; e < 8; e++) {
                float v = fmaxf(fmaxf(s[0][e], s[1][e]), fmaxf(s[2][e], s[3][e]));
                for (int off = 8; off; off >>= 1)
                    v = fmaxf(v, __shfl_xor(v, off, 32));
                float mn = fmaxf(m[e], v);
                alpha[e] = exp2f(m[e] - mn);
                m[e] = mn;
            }
            float rs[8];
            for (int e = 0; e < 8; e++) rs[e] = 0.f;
            for (int t = 0; t < 4; t++)
                for (int e = 0; e < 8; e++) {
                    float p = exp2f(s[t][e] - m[e]);
                    s[t][e] = p;
                    rs[e] += p;
                }
            for (int e = 0; e < 8; e++) {
                float v = rs[e];
                for (int off = 8; off; off >>= 1) v += __shfl_xor(v, off, 32);
                l[e] = l[e] * alpha[e] + v;
            }
            for (int n = 0; n < 4; n++)
                for (int e = 0; e < 8; e++) o[n][e] *= alpha[e];

            // ---- stage P to per-wave LDS, re-read as A fragments ----
            for (int t = 0; t < 4; t++)
                for (int e = 0; e < 8; e++)
                    Ps[wave][laneHi * 8 + e][t * 16 + laneLo] = (bf16)s[t][e];
            BF16x16 pf[2];
            {
                const bf16* pr = &Ps[wave][laneLo][0];
                pf[0].q[0] = *(const uint4*)(pr + laneHi * 8);
                pf[0].q[1] = *(const uint4*)(pr + 16 + laneHi * 8);
                pf[1].q[0] = *(const uint4*)(pr + 32 + laneHi * 8);
                pf[1].q[1] = *(const uint4*)(pr + 48 + laneHi * 8);
            }
            // ---- O += P V (4 dh tiles x 2 key-chunks) ----
            for (int n = 0; n < 4; n++) {
                int c = n * 16 + laneLo;     // dh column
                BF16x16 vf0, vf1;            // B frag: lane=dh col, elems=key
                vf0.q[0] = *(const uint4*)&Vt[c][laneHi * 8];
                vf0.q[1] = *(const uint4*)&Vt[c][16 + laneHi * 8];
                vf1.q[0] = *(const uint4*)&Vt[c][32 + laneHi * 8];
                vf1.q[1] = *(const uint4*)&Vt[c][48 + laneHi * 8];
                o[n] = __builtin_amdgcn_wmma_f32_16x16x32_bf16(
                           false, pf[0].v, false, vf0.v, (short)0, o[n], false, false);
                o[n] = __builtin_amdgcn_wmma_f32_16x16x32_bf16(
                           false, pf[1].v, false, vf1.v, (short)0, o[n], false, false);
            }
        }
    }

    // Epilogue: normalize, write bf16 attn output in [B,S,D] layout
    const int b = bh >> 4, h = bh & (HH - 1);
    for (int e = 0; e < 8; e++) {
        float inv = 1.0f / l[e];
        int row = qtile + laneHi * 8 + e;
        for (int n = 0; n < 4; n++) {
            int dh = n * 16 + laneLo;
            Outp[((long)(b * SS + row)) * DD + h * DHD + dh] = (bf16)(o[n][e] * inv);
        }
    }
}

// ---------------------------------------------------------------------------
// Host-side launcher
// ---------------------------------------------------------------------------
extern "C" void kernel_launch(void* const* d_in, const int* in_sizes, int n_in,
                              void* d_out, int out_size, void* d_ws, size_t ws_size,
                              hipStream_t stream) {
    (void)in_sizes; (void)n_in; (void)out_size; (void)ws_size;
    const float* x  = (const float*)d_in[0];
    const float* Wq = (const float*)d_in[1];
    const float* bq = (const float*)d_in[2];
    const float* Wk = (const float*)d_in[3];
    const float* bk = (const float*)d_in[4];
    const float* Wv = (const float*)d_in[5];
    const float* bv = (const float*)d_in[6];
    const float* Wo = (const float*)d_in[7];
    const float* bo = (const float*)d_in[8];
    float* out = (float*)d_out;

    char* ws = (char*)d_ws;
    size_t off = 0;
    auto take = [&](size_t elems) { void* p = ws + off; off += elems * 2; return p; };
    bf16* xb   = (bf16*)take((size_t)MM * DD);
    bf16* Wqb  = (bf16*)take((size_t)DD * DD);
    bf16* Wkb  = (bf16*)take((size_t)DD * DD);
    bf16* Wvb  = (bf16*)take((size_t)DD * DD);
    bf16* Wob  = (bf16*)take((size_t)DD * DD);
    bf16* Qh   = (bf16*)take((size_t)MM * DD);
    bf16* Kh   = (bf16*)take((size_t)MM * DD);
    bf16* Vh   = (bf16*)take((size_t)MM * DD);
    bf16* attn = (bf16*)take((size_t)MM * DD);

    // 1) convert x to bf16
    convert_bf16_kernel<<<(MM * DD) / (256 * 4), 256, 0, stream>>>(x, xb, (long)MM * DD);
    // 2) transpose+convert weights to [N][K] bf16
    dim3 tb(32, 8), tg(DD / 32, DD / 32);
    transpose_convert_kernel<<<tg, tb, 0, stream>>>(Wq, Wqb);
    transpose_convert_kernel<<<tg, tb, 0, stream>>>(Wk, Wkb);
    transpose_convert_kernel<<<tg, tb, 0, stream>>>(Wv, Wvb);
    transpose_convert_kernel<<<tg, tb, 0, stream>>>(Wo, Wob);
    // 3) QKV projections (Q pre-scaled by 1/sqrt(Dh) * log2(e))
    dim3 gg(MM / 128, DD / 128);
    const float qscale = 0.125f * 1.4426950408889634f;
    gemm_bf16_kernel<0><<<gg, 256, 0, stream>>>(xb, Wqb, bq, Qh, qscale);
    gemm_bf16_kernel<0><<<gg, 256, 0, stream>>>(xb, Wkb, bk, Kh, 1.0f);
    gemm_bf16_kernel<0><<<gg, 256, 0, stream>>>(xb, Wvb, bv, Vh, 1.0f);
    // 4) causal flash attention
    attn_kernel<<<dim3(SS / 128, BB * HH), 256, 0, stream>>>(Qh, Kh, Vh, attn);
    // 5) output projection -> fp32
    gemm_bf16_kernel<1><<<gg, 256, 0, stream>>>(attn, Wob, bo, out, 1.0f);
}
